// Ligand_Intra_Bond_Layer_62672162783756
// MI455X (gfx1250) — matrix operand order repaired and checked by reading, not compiled, not run on hardware
//
#include <hip/hip_runtime.h>

// ---------------------------------------------------------------------------
// Types for CDNA5 WMMA (wave32): v_wmma_f32_16x16x32_bf16
// ---------------------------------------------------------------------------
typedef __bf16 bf16_t;
typedef __attribute__((ext_vector_type(16))) __bf16 v16bf;
typedef __attribute__((ext_vector_type(8)))  __bf16 v8bf;
typedef __attribute__((ext_vector_type(8)))  float  v8f;

__device__ __forceinline__ v8f vzero() { v8f z = {}; return z; }

// A fragment (16-bit 16x32, ISA 7.12.2): lane half h = lane>>4, row m = lane&15.
// Elements 0..7  = A[m][h*8 .. h*8+7], elements 8..15 = A[m][16+h*8 .. 16+h*8+7]
// -> two contiguous 16B runs in row-major LDS (lda must be multiple of 8).
__device__ __forceinline__ v16bf load_a_frag(const bf16_t* As, int lda, int lane) {
  const int half = lane >> 4, m = lane & 15;
  const bf16_t* p = As + m * lda + half * 8;
  v8bf lo = *(const v8bf*)p;
  v8bf hi = *(const v8bf*)(p + 16);
  return __builtin_shufflevector(lo, hi, 0, 1, 2, 3, 4, 5, 6, 7,
                                 8, 9, 10, 11, 12, 13, 14, 15);
}

// B fragment: column n = lane&15 (+ncol), K = half*8 + 0..7, 16 + half*8 + 0..7.
// Weights are pre-packed as [kchunk][n][32 k] so this is two contiguous 16B runs.
__device__ __forceinline__ v16bf load_b_frag(const bf16_t* Bchunk, int ncol, int lane) {
  const int half = lane >> 4, n = lane & 15;
  const bf16_t* p = Bchunk + ((ncol + n) << 5) + half * 8;
  v8bf lo = *(const v8bf*)p;
  v8bf hi = *(const v8bf*)(p + 16);
  return __builtin_shufflevector(lo, hi, 0, 1, 2, 3, 4, 5, 6, 7,
                                 8, 9, 10, 11, 12, 13, 14, 15);
}

__device__ __forceinline__ v8f wmma_acc(v16bf a, v16bf b, v8f acc) {
  return __builtin_amdgcn_wmma_f32_16x16x32_bf16(
      false, a, false, b, (short)0, acc, false, false);
}

// Copy one packed weight chunk (32 x Nw, bf16) global -> LDS with 128b ops.
__device__ __forceinline__ void stage_chunk(bf16_t* sB, const bf16_t* Wp,
                                            int ck, int Nw, int tid) {
  const uint4* s = (const uint4*)(Wp + (size_t)ck * Nw * 32);
  uint4* d = (uint4*)sB;
  for (int i = tid; i < Nw * 4; i += 128) d[i] = s[i];
}

// 4x fp32 -> 4x bf16 packed 8-byte store
__device__ __forceinline__ void cvt4(bf16_t* d, float4 f) {
  union { bf16_t b[4]; unsigned long long u; } t;
  t.b[0] = (bf16_t)f.x; t.b[1] = (bf16_t)f.y;
  t.b[2] = (bf16_t)f.z; t.b[3] = (bf16_t)f.w;
  *(unsigned long long*)d = t.u;
}

// ---------------------------------------------------------------------------
// Problem constants
// ---------------------------------------------------------------------------
#define NN 10000
#define EE 320000

// ---------------------------------------------------------------------------
// K0: fp32 weights -> bf16 packed in WMMA-B layout: d[((k>>5)*Nw + n)*32 + (k&31)]
// ---------------------------------------------------------------------------
__global__ void pack_b_kernel(const float* __restrict__ s, bf16_t* __restrict__ d,
                              int K, int Nw) {
  int i = blockIdx.x * 256 + threadIdx.x;
  if (i < K * Nw) {
    int k = i / Nw, n = i - k * Nw;
    d[((((k >> 5) * Nw) + n) << 5) + (k & 31)] = (bf16_t)s[i];
  }
}

// ---------------------------------------------------------------------------
// K1: hn3[N,384] = h @ [w_nll | w_nlr | w_nl] + biases   (fp32 out for gathers)
// grid = (N/16, 6); block = 128 (4 waves); blockIdx.y selects 64-col macrotile
// ---------------------------------------------------------------------------
__global__ __launch_bounds__(128) void node_gemm_kernel(
    const float* __restrict__ h,
    const bf16_t* __restrict__ w_nll, const bf16_t* __restrict__ w_nlr,
    const bf16_t* __restrict__ w_nl,
    const float* __restrict__ b_nll, const float* __restrict__ b_nlr,
    const float* __restrict__ b_nl, float* __restrict__ hn3) {
  __shared__ alignas(32) bf16_t sA[16 * 128];
  __shared__ alignas(32) bf16_t sB[32 * 64];
  const int tid = threadIdx.x, wave = tid >> 5, lane = tid & 31;
  const int half = lane >> 4, n16 = lane & 15;
  const int r0 = blockIdx.x * 16;
  const int y = blockIdx.y;
  const bf16_t* W = (y < 2) ? w_nll : (y < 4) ? w_nlr : w_nl;
  const float*  B = (y < 2) ? b_nll : (y < 4) ? b_nlr : b_nl;
  const int c0 = (y & 1) * 64;

  for (int i = tid; i < 16 * 32; i += 128) {
    int r = i >> 5, c4 = (i & 31) * 4;
    cvt4(&sA[r * 128 + c4], *(const float4*)&h[(r0 + r) * 128 + c4]);
  }

  v8f acc = vzero();
  for (int k0 = 0; k0 < 128; k0 += 32) {
    __syncthreads();
    {  // stage packed columns [c0 .. c0+63] of this 32-k chunk (contiguous)
      const uint4* s = (const uint4*)(W + ((size_t)(k0 >> 5) * 128 + c0) * 32);
      uint4* d = (uint4*)sB;
      for (int i = tid; i < 64 * 4; i += 128) d[i] = s[i];
    }
    __syncthreads();
    v16bf a = load_a_frag(sA + k0, 128, lane);
    acc = wmma_acc(a, load_b_frag(sB, wave * 16, lane), acc);
  }
  const int gcol = y * 64 + wave * 16 + n16;   // column in 384-wide hn3
  const int wcol = c0 + wave * 16 + n16;       // column within this weight
#pragma unroll
  for (int r = 0; r < 8; ++r)
    hn3[(r0 + r + 8 * half) * 384 + gcol] = acc[r] + B[wcol];
}

// ---------------------------------------------------------------------------
// K2: ea[E,64] = LN(cat[edge_attr, time]) @ w_ee + b_ee
// grid = E/64; block = 128; wave w owns rows 16w..16w+15, computes 4 N-tiles
// ---------------------------------------------------------------------------
__global__ __launch_bounds__(128) void edge_emb_kernel(
    const float* __restrict__ edge_attr, const float* __restrict__ edge_time,
    const float* __restrict__ g_en, const float* __restrict__ bt_en,
    const bf16_t* __restrict__ w_ee, const float* __restrict__ b_ee,
    float* __restrict__ ea_out) {
  __shared__ alignas(16) float  sx[64 * 96];
  __shared__ alignas(32) bf16_t sA[64 * 96];
  __shared__ alignas(32) bf16_t sB[32 * 64];
  const int tid = threadIdx.x, wave = tid >> 5, lane = tid & 31;
  const int half = lane >> 4, n16 = lane & 15;
  const int e0 = blockIdx.x * 64;
  const int rbase = wave * 16;

  for (int i = tid; i < 64 * 16; i += 128) {
    int r = i >> 4, c4 = (i & 15) * 4;
    *(float4*)&sx[r * 96 + c4] = *(const float4*)&edge_attr[(e0 + r) * 64 + c4];
  }
  for (int i = tid; i < 64 * 8; i += 128) {
    int r = i >> 3, c4 = (i & 7) * 4;
    *(float4*)&sx[r * 96 + 64 + c4] = *(const float4*)&edge_time[(e0 + r) * 32 + c4];
  }
  __syncthreads();

  if (lane < 16) {  // each wave LayerNorms its own 16 rows (one row per lane)
    int row = rbase + lane;
    float m = 0.f, v = 0.f;
    for (int c = 0; c < 96; ++c) m += sx[row * 96 + c];
    m *= (1.f / 96.f);
    for (int c = 0; c < 96; ++c) { float d = sx[row * 96 + c] - m; v += d * d; }
    v *= (1.f / 96.f);
    float rs = rsqrtf(v + 1e-5f);
    for (int c = 0; c < 96; ++c)
      sA[row * 96 + c] = (bf16_t)((sx[row * 96 + c] - m) * rs * g_en[c] + bt_en[c]);
  }

  v8f acc[4];
#pragma unroll
  for (int t = 0; t < 4; ++t) acc[t] = vzero();
  for (int k0 = 0; k0 < 96; k0 += 32) {
    __syncthreads();
    stage_chunk(sB, w_ee, k0 >> 5, 64, tid);
    __syncthreads();
    v16bf a = load_a_frag(sA + rbase * 96 + k0, 96, lane);
#pragma unroll
    for (int t = 0; t < 4; ++t)
      acc[t] = wmma_acc(a, load_b_frag(sB, t * 16, lane), acc[t]);
  }
#pragma unroll
  for (int t = 0; t < 4; ++t) {
    int col = t * 16 + n16;
#pragma unroll
    for (int r = 0; r < 8; ++r)
      ea_out[(e0 + rbase + r + 8 * half) * 64 + col] = acc[t][r] + b_ee[col];
  }
}

// ---------------------------------------------------------------------------
// K3: per-edge left/right MLP with fused gather + LN + SiLU + atomic scatter
// grid = E/64; block = 128; wave w owns rows 16w..16w+15, computes 8 N-tiles
// ---------------------------------------------------------------------------
__global__ __launch_bounds__(128) void edge_mlp_kernel(
    const float* __restrict__ ea, const int* __restrict__ src,
    const int* __restrict__ dst, const float* __restrict__ hn3,
    const bf16_t* __restrict__ w_bll, const float* __restrict__ b_bll,
    const bf16_t* __restrict__ w_blr, const float* __restrict__ b_blr,
    const bf16_t* __restrict__ w1l, const float* __restrict__ b1l,
    const float* __restrict__ gl, const float* __restrict__ btl,
    const bf16_t* __restrict__ w2l, const float* __restrict__ b2l,
    const bf16_t* __restrict__ w1r, const float* __restrict__ b1r,
    const float* __restrict__ gr, const float* __restrict__ btr,
    const bf16_t* __restrict__ w2r, const float* __restrict__ b2r,
    float* __restrict__ acc_i, float* __restrict__ acc_j) {
  __shared__ alignas(32) bf16_t sEA[64 * 64];    //  8KB
  __shared__ alignas(32) bf16_t sB[32 * 128];    //  8KB
  __shared__ alignas(16) float  sC[64 * 128];    // 32KB
  __shared__ alignas(32) bf16_t sX[64 * 256];    // 32KB
  const int tid = threadIdx.x, wave = tid >> 5, lane = tid & 31;
  const int half = lane >> 4, n16 = lane & 15;
  const int e0 = blockIdx.x * 64;
  const int rbase = wave * 16;

  for (int i = tid; i < 64 * 16; i += 128)
    cvt4(&sEA[i * 4], *(const float4*)&ea[e0 * 64 + i * 4]);

  for (int side = 0; side < 2; ++side) {
    const bf16_t* wB = side ? w_blr : w_bll;
    const float*  bB = side ? b_blr : b_bll;
    const bf16_t* w1 = side ? w1r : w1l;
    const float*  b1 = side ? b1r : b1l;
    const float*  g  = side ? gr : gl;
    const float*  bt = side ? btr : btl;
    const bf16_t* w2 = side ? w2r : w2l;
    const float*  b2 = side ? b2r : b2l;
    const int*    idx = side ? dst : src;
    float*        accp = side ? acc_j : acc_i;
    const int     hoff = side ? 128 : 0;

    // edgeX = sEA(64x64) @ wB(64x128) + bB  -> sC
    v8f acc[8];
#pragma unroll
    for (int t = 0; t < 8; ++t) acc[t] = vzero();
    for (int k0 = 0; k0 < 64; k0 += 32) {
      __syncthreads();
      stage_chunk(sB, wB, k0 >> 5, 128, tid);
      __syncthreads();
      v16bf a = load_a_frag(sEA + rbase * 64 + k0, 64, lane);
#pragma unroll
      for (int t = 0; t < 8; ++t)
        acc[t] = wmma_acc(a, load_b_frag(sB, t * 16, lane), acc[t]);
    }
    __syncthreads();
#pragma unroll
    for (int t = 0; t < 8; ++t) {
      int col = t * 16 + n16;
#pragma unroll
      for (int r = 0; r < 8; ++r)
        sC[(rbase + r + 8 * half) * 128 + col] = acc[t][r] + bB[col];
    }
    __syncthreads();

    // cat[edgeX, gathered node features] -> sX bf16 [64 x 256]
    for (int i = tid; i < 64 * 32; i += 128) {
      int r = i >> 5, c4 = (i & 31) * 4;
      cvt4(&sX[r * 256 + c4], *(const float4*)&sC[r * 128 + c4]);
      cvt4(&sX[r * 256 + 128 + c4],
           *(const float4*)&hn3[idx[e0 + r] * 384 + hoff + c4]);
    }

    // layer1: sX(64x256) @ w1(256x128) + b1 -> sC
#pragma unroll
    for (int t = 0; t < 8; ++t) acc[t] = vzero();
    for (int k0 = 0; k0 < 256; k0 += 32) {
      __syncthreads();
      stage_chunk(sB, w1, k0 >> 5, 128, tid);
      __syncthreads();
      v16bf a = load_a_frag(sX + rbase * 256 + k0, 256, lane);
#pragma unroll
      for (int t = 0; t < 8; ++t)
        acc[t] = wmma_acc(a, load_b_frag(sB, t * 16, lane), acc[t]);
    }
    __syncthreads();
#pragma unroll
    for (int t = 0; t < 8; ++t) {
      int col = t * 16 + n16;
#pragma unroll
      for (int r = 0; r < 8; ++r)
        sC[(rbase + r + 8 * half) * 128 + col] = acc[t][r] + b1[col];
    }
    __syncthreads();

    // LN(128) + SiLU (per-wave rows) -> bf16 into sX (reused as 64x128, ld=128)
    if (lane < 16) {
      int row = rbase + lane;
      float m = 0.f, v = 0.f;
      for (int c = 0; c < 128; ++c) m += sC[row * 128 + c];
      m *= (1.f / 128.f);
      for (int c = 0; c < 128; ++c) { float d = sC[row * 128 + c] - m; v += d * d; }
      v *= (1.f / 128.f);
      float rs = rsqrtf(v + 1e-5f);
      for (int c = 0; c < 128; ++c) {
        float y = (sC[row * 128 + c] - m) * rs * g[c] + bt[c];
        y = y / (1.f + __expf(-y));
        sX[row * 128 + c] = (bf16_t)y;
      }
    }
    __syncthreads();

    // layer2: (64x128) @ w2(128x128) + b2 -> atomic scatter to nodes
#pragma unroll
    for (int t = 0; t < 8; ++t) acc[t] = vzero();
    for (int k0 = 0; k0 < 128; k0 += 32) {
      __syncthreads();
      stage_chunk(sB, w2, k0 >> 5, 128, tid);
      __syncthreads();
      v16bf a = load_a_frag(sX + rbase * 128 + k0, 128, lane);
#pragma unroll
      for (int t = 0; t < 8; ++t)
        acc[t] = wmma_acc(a, load_b_frag(sB, t * 16, lane), acc[t]);
    }
#pragma unroll
    for (int r = 0; r < 8; ++r) {
      int node = idx[e0 + rbase + r + 8 * half];
#pragma unroll
      for (int t = 0; t < 8; ++t) {
        int col = t * 16 + n16;
        atomicAdd(&accp[node * 128 + col], acc[t][r] + b2[col]);
      }
    }
    __syncthreads();
  }
}

// ---------------------------------------------------------------------------
// K4: h_bond = LN384(cat[acc_i[src]+acc_j[dst], hnode[src]+hnode[dst], ea@w_bl])
//     out    = SiLU(h_bond @ w_o1 + b_o1) @ w_o2 + b_o2
// grid = E/64; block = 128; wave w owns rows 16w..16w+15
// ---------------------------------------------------------------------------
__global__ __launch_bounds__(128) void bond_out_kernel(
    const float* __restrict__ ea, const int* __restrict__ src,
    const int* __restrict__ dst, const float* __restrict__ hn3,
    const float* __restrict__ acc_i, const float* __restrict__ acc_j,
    const bf16_t* __restrict__ w_bl, const float* __restrict__ b_bl,
    const float* __restrict__ g_ln, const float* __restrict__ bt_ln,
    const bf16_t* __restrict__ w_o1, const float* __restrict__ b_o1,
    const bf16_t* __restrict__ w_o2, const float* __restrict__ b_o2,
    float* __restrict__ out) {
  __shared__ alignas(32) bf16_t sEA[64 * 64];    //  8KB
  __shared__ alignas(32) bf16_t sB[32 * 128];    //  8KB
  __shared__ alignas(32) bf16_t sHB[64 * 384];   // 48KB (concat kept in bf16)
  __shared__ alignas(32) bf16_t sY[64 * 128];    // 16KB
  const int tid = threadIdx.x, wave = tid >> 5, lane = tid & 31;
  const int half = lane >> 4, n16 = lane & 15;
  const int e0 = blockIdx.x * 64;
  const int rbase = wave * 16;

  for (int i = tid; i < 64 * 16; i += 128)
    cvt4(&sEA[i * 4], *(const float4*)&ea[e0 * 64 + i * 4]);
  for (int i = tid; i < 64 * 32; i += 128) {
    int r = i >> 5, c4 = (i & 31) * 4;
    int s = src[e0 + r], d = dst[e0 + r];
    float4 ai = *(const float4*)&acc_i[s * 128 + c4];
    float4 aj = *(const float4*)&acc_j[d * 128 + c4];
    cvt4(&sHB[r * 384 + c4],
         make_float4(ai.x + aj.x, ai.y + aj.y, ai.z + aj.z, ai.w + aj.w));
    float4 hs = *(const float4*)&hn3[s * 384 + 256 + c4];
    float4 hd = *(const float4*)&hn3[d * 384 + 256 + c4];
    cvt4(&sHB[r * 384 + 128 + c4],
         make_float4(hs.x + hd.x, hs.y + hd.y, hs.z + hd.z, hs.w + hd.w));
  }

  // ea @ w_bl (64 -> 128) + b_bl into sHB[:,256:384]
  v8f acc[8];
#pragma unroll
  for (int t = 0; t < 8; ++t) acc[t] = vzero();
  for (int k0 = 0; k0 < 64; k0 += 32) {
    __syncthreads();
    stage_chunk(sB, w_bl, k0 >> 5, 128, tid);
    __syncthreads();
    v16bf a = load_a_frag(sEA + rbase * 64 + k0, 64, lane);
#pragma unroll
    for (int t = 0; t < 8; ++t)
      acc[t] = wmma_acc(a, load_b_frag(sB, t * 16, lane), acc[t]);
  }
  __syncthreads();
#pragma unroll
  for (int t = 0; t < 8; ++t) {
    int col = t * 16 + n16;
#pragma unroll
    for (int r = 0; r < 8; ++r)
      sHB[(rbase + r + 8 * half) * 384 + 256 + col] = (bf16_t)(acc[t][r] + b_bl[col]);
  }
  __syncthreads();

  // LN(384) in place (per-wave rows, stats from bf16)
  if (lane < 16) {
    int row = rbase + lane;
    float m = 0.f, v = 0.f;
    for (int c = 0; c < 384; ++c) m += (float)sHB[row * 384 + c];
    m *= (1.f / 384.f);
    for (int c = 0; c < 384; ++c) {
      float d = (float)sHB[row * 384 + c] - m; v += d * d;
    }
    v *= (1.f / 384.f);
    float rs = rsqrtf(v + 1e-5f);
    for (int c = 0; c < 384; ++c)
      sHB[row * 384 + c] =
          (bf16_t)(((float)sHB[row * 384 + c] - m) * rs * g_ln[c] + bt_ln[c]);
  }
  __syncthreads();

  // o1: (64x384) @ w_o1(384x128) + b_o1, SiLU -> sY bf16
#pragma unroll
  for (int t = 0; t < 8; ++t) acc[t] = vzero();
  for (int k0 = 0; k0 < 384; k0 += 32) {
    __syncthreads();
    stage_chunk(sB, w_o1, k0 >> 5, 128, tid);
    __syncthreads();
    v16bf a = load_a_frag(sHB + rbase * 384 + k0, 384, lane);
#pragma unroll
    for (int t = 0; t < 8; ++t)
      acc[t] = wmma_acc(a, load_b_frag(sB, t * 16, lane), acc[t]);
  }
  __syncthreads();
#pragma unroll
  for (int t = 0; t < 8; ++t) {
    int col = t * 16 + n16;
#pragma unroll
    for (int r = 0; r < 8; ++r) {
      float y = acc[t][r] + b_o1[col];
      sY[(rbase + r + 8 * half) * 128 + col] = (bf16_t)(y / (1.f + __expf(-y)));
    }
  }
  __syncthreads();

  // o2: (64x128) @ w_o2(128x64) + b_o2 -> out
  v8f acc2[4];
#pragma unroll
  for (int t = 0; t < 4; ++t) acc2[t] = vzero();
  for (int k0 = 0; k0 < 128; k0 += 32) {
    __syncthreads();
    stage_chunk(sB, w_o2, k0 >> 5, 64, tid);
    __syncthreads();
    v16bf a = load_a_frag(sY + rbase * 128 + k0, 128, lane);
#pragma unroll
    for (int t = 0; t < 4; ++t)
      acc2[t] = wmma_acc(a, load_b_frag(sB, t * 16, lane), acc2[t]);
  }
#pragma unroll
  for (int t = 0; t < 4; ++t) {
    int col = t * 16 + n16;
#pragma unroll
    for (int r = 0; r < 8; ++r)
      out[(e0 + rbase + r + 8 * half) * 64 + col] = acc2[t][r] + b_o2[col];
  }
}

// ---------------------------------------------------------------------------
// Host launcher
// ---------------------------------------------------------------------------
extern "C" void kernel_launch(void* const* d_in, const int* in_sizes, int n_in,
                              void* d_out, int out_size, void* d_ws, size_t ws_size,
                              hipStream_t stream) {
  (void)in_sizes; (void)n_in; (void)out_size; (void)ws_size;
  const float* h         = (const float*)d_in[0];
  const float* edge_attr = (const float*)d_in[1];
  const float* edge_time = (const float*)d_in[2];
  const int*   eidx      = (const int*)d_in[3];
  const float* w_ee  = (const float*)d_in[4];  const float* b_ee  = (const float*)d_in[5];
  const float* g_en  = (const float*)d_in[6];  const float* bt_en = (const float*)d_in[7];
  const float* w_nl  = (const float*)d_in[8];  const float* b_nl  = (const float*)d_in[9];
  const float* w_bl  = (const float*)d_in[10]; const float* b_bl  = (const float*)d_in[11];
  const float* w_bll = (const float*)d_in[12]; const float* b_bll = (const float*)d_in[13];
  const float* w_blr = (const float*)d_in[14]; const float* b_blr = (const float*)d_in[15];
  const float* w_nll = (const float*)d_in[16]; const float* b_nll = (const float*)d_in[17];
  const float* w_nlr = (const float*)d_in[18]; const float* b_nlr = (const float*)d_in[19];
  const float* w1l = (const float*)d_in[20]; const float* b1l = (const float*)d_in[21];
  const float* gl  = (const float*)d_in[22]; const float* btl = (const float*)d_in[23];
  const float* w2l = (const float*)d_in[24]; const float* b2l = (const float*)d_in[25];
  const float* w1r = (const float*)d_in[26]; const float* b1r = (const float*)d_in[27];
  const float* gr  = (const float*)d_in[28]; const float* btr = (const float*)d_in[29];
  const float* w2r = (const float*)d_in[30]; const float* b2r = (const float*)d_in[31];
  const float* g_ln = (const float*)d_in[32]; const float* bt_ln = (const float*)d_in[33];
  const float* w_o1 = (const float*)d_in[34]; const float* b_o1 = (const float*)d_in[35];
  const float* w_o2 = (const float*)d_in[36]; const float* b_o2 = (const float*)d_in[37];
  const int* src = eidx;
  const int* dst = eidx + EE;

  char* ws = (char*)d_ws;
  size_t off = 0;
  auto alloc = [&](size_t bytes) -> void* {
    off = (off + 255) & ~(size_t)255;
    void* p = ws + off;
    off += bytes;
    return p;
  };
  bf16_t* wee_b  = (bf16_t*)alloc(96 * 64 * 2);
  bf16_t* wnll_b = (bf16_t*)alloc(128 * 128 * 2);
  bf16_t* wnlr_b = (bf16_t*)alloc(128 * 128 * 2);
  bf16_t* wnl_b  = (bf16_t*)alloc(128 * 128 * 2);
  bf16_t* wbll_b = (bf16_t*)alloc(64 * 128 * 2);
  bf16_t* wblr_b = (bf16_t*)alloc(64 * 128 * 2);
  bf16_t* wbl_b  = (bf16_t*)alloc(64 * 128 * 2);
  bf16_t* w1l_b  = (bf16_t*)alloc(256 * 128 * 2);
  bf16_t* w2l_b  = (bf16_t*)alloc(128 * 128 * 2);
  bf16_t* w1r_b  = (bf16_t*)alloc(256 * 128 * 2);
  bf16_t* w2r_b  = (bf16_t*)alloc(128 * 128 * 2);
  bf16_t* wo1_b  = (bf16_t*)alloc(384 * 128 * 2);
  bf16_t* wo2_b  = (bf16_t*)alloc(128 * 64 * 2);
  float* hn3  = (float*)alloc((size_t)NN * 384 * 4);
  float* ea   = (float*)alloc((size_t)EE * 64 * 4);
  float* acci = (float*)alloc((size_t)NN * 128 * 4);
  float* accj = (float*)alloc((size_t)NN * 128 * 4);

  auto pack = [&](const float* s, bf16_t* d, int K, int Nw) {
    pack_b_kernel<<<(K * Nw + 255) / 256, 256, 0, stream>>>(s, d, K, Nw);
  };
  pack(w_ee,  wee_b,  96, 64);
  pack(w_nll, wnll_b, 128, 128);
  pack(w_nlr, wnlr_b, 128, 128);
  pack(w_nl,  wnl_b,  128, 128);
  pack(w_bll, wbll_b, 64, 128);
  pack(w_blr, wblr_b, 64, 128);
  pack(w_bl,  wbl_b,  64, 128);
  pack(w1l,   w1l_b,  256, 128);
  pack(w2l,   w2l_b,  128, 128);
  pack(w1r,   w1r_b,  256, 128);
  pack(w2r,   w2r_b,  128, 128);
  pack(w_o1,  wo1_b,  384, 128);
  pack(w_o2,  wo2_b,  128, 64);

  hipMemsetAsync(acci, 0, (size_t)NN * 128 * 4, stream);
  hipMemsetAsync(accj, 0, (size_t)NN * 128 * 4, stream);

  node_gemm_kernel<<<dim3(NN / 16, 6), 128, 0, stream>>>(
      h, wnll_b, wnlr_b, wnl_b, b_nll, b_nlr, b_nl, hn3);

  edge_emb_kernel<<<EE / 64, 128, 0, stream>>>(
      edge_attr, edge_time, g_en, bt_en, wee_b, b_ee, ea);

  edge_mlp_kernel<<<EE / 64, 128, 0, stream>>>(
      ea, src, dst, hn3,
      wbll_b, b_bll, wblr_b, b_blr,
      w1l_b, b1l, gl, btl, w2l_b, b2l,
      w1r_b, b1r, gr, btr, w2r_b, b2r,
      acci, accj);

  bond_out_kernel<<<EE / 64, 128, 0, stream>>>(
      ea, src, dst, hn3, acci, accj,
      wbl_b, b_bl, g_ln, bt_ln, wo1_b, b_o1, wo2_b, b_o2,
      (float*)d_out);
}
